// EncoderLayer_61624190763290
// MI455X (gfx1250) — matrix operand order repaired and checked by reading, not compiled, not run on hardware
//
#include <hip/hip_runtime.h>
#include <hip/hip_bf16.h>
#include <math.h>

// ---------------------------------------------------------------------------
// Problem constants (from reference)
// ---------------------------------------------------------------------------
constexpr int F  = 512;
constexpr int N  = 32;
constexpr int C  = 256;
constexpr int H  = 8;
constexpr int E  = 32;      // C / H
constexpr int MP = 128;
constexpr float EPS = 1e-5f;

// CDNA5 async global->LDS path (probe-guarded: falls back to sync LDS copy)
#if defined(__gfx1250__) && __has_builtin(__builtin_amdgcn_global_load_async_to_lds_b128) && __has_builtin(__builtin_amdgcn_s_wait_asynccnt)
#define HAVE_ASYNC_LDS 1
#else
#define HAVE_ASYNC_LDS 0
#endif

#if HAVE_ASYNC_LDS
typedef int v4i_vs __attribute__((vector_size(16)));
typedef __attribute__((address_space(1))) v4i_vs gv4i;   // global int4*
typedef __attribute__((address_space(3))) v4i_vs sv4i;   // LDS int4*
#endif

// ---------------------------------------------------------------------------
// WMMA fragment types / helpers (CDNA5, wave32, 16x16x32 bf16)
// ---------------------------------------------------------------------------
typedef __attribute__((ext_vector_type(16))) __bf16 v16bf;
typedef __attribute__((ext_vector_type(8)))  float  v8f;

__device__ __forceinline__ __bf16 tobf(float f) { return (__bf16)f; }

// A matrix 16x32 (MxK), bf16 layout per ISA 7.12.2:
// lane: m = lane&15, half = lane>>4 ; VGPR j holds K = (j>>2)*16 + half*8 + (j&3)*2 (+1)
template <typename SRC>
__device__ __forceinline__ v16bf load_frag_a(SRC src, int lane) {
    v16bf a;
    const int m = lane & 15, half = lane >> 4;
#pragma unroll
    for (int j = 0; j < 8; ++j) {
        const int kk = ((j >> 2) << 4) + (half << 3) + ((j & 3) << 1);
        a[2 * j]     = tobf(src(m, kk));
        a[2 * j + 1] = tobf(src(m, kk + 1));
    }
    return a;
}

// B matrix 32x16 (KxN), bf16: lane: n = lane&15, half = lane>>4 ;
// VGPR j holds K = (j>>2)*16 + half*8 + (j&3)*2 (+1)
template <typename SRC>
__device__ __forceinline__ v16bf load_frag_b(SRC src, int lane) {
    v16bf b;
    const int n = lane & 15, half = lane >> 4;
#pragma unroll
    for (int j = 0; j < 8; ++j) {
        const int kk = ((j >> 2) << 4) + (half << 3) + ((j & 3) << 1);
        b[2 * j]     = tobf(src(kk, n));
        b[2 * j + 1] = tobf(src(kk + 1, n));
    }
    return b;
}

__device__ __forceinline__ v8f wmma_bf16(v16bf a, v16bf b, v8f c) {
    return __builtin_amdgcn_wmma_f32_16x16x32_bf16(false, a, false, b, (short)0, c,
                                                   false, false);
}

__device__ __forceinline__ float gelu_exact(float x) {
    return 0.5f * x * (1.0f + erff(x * 0.70710678118654752f));
}

// ---------------------------------------------------------------------------
// Generic GEMM: Y[M,Nc] = act(X[M,K] @ W[K,Nc] + bias).
// Block = 8 waves stacked in M over ONE shared 16-wide column tile; the 32x16
// W panel is staged into LDS once per K-step via async global->LDS DMA
// (ASYNCcnt path), so W global traffic drops 8x.
// ---------------------------------------------------------------------------
__global__ __launch_bounds__(256) void gemm_bias_act_kernel(
    const float* __restrict__ X, const float* __restrict__ W,
    const float* __restrict__ bias, float* __restrict__ Y,
    int M, int K, int Nc, int act) {
    __shared__ float w_lds[32 * 16];    // K-step panel of W, fp32

    const int tid  = threadIdx.x;
    const int lane = tid & 31;
    const int wave = tid >> 5;
    const int tilesN = Nc >> 4;
    const int mBlock = blockIdx.x / tilesN;
    const int tn     = blockIdx.x % tilesN;
    const int rowB   = mBlock * 128 + wave * 16;
    const int colB   = tn << 4;
    if (rowB >= M) return;

    __builtin_prefetch(X + (size_t)rowB * K, 0, 3);  // global_prefetch_b8 (CDNA5)

    v8f acc = {};
    for (int kb = 0; kb < K; kb += 32) {
        // ---- stage W[kb:kb+32, colB:colB+16] into LDS -----------------------
        if (tid < 128) {
            const int kk = tid >> 2, ng = (tid & 3) << 2;  // 4 floats per lane
#if HAVE_ASYNC_LDS
            __builtin_amdgcn_global_load_async_to_lds_b128(
                (gv4i*)(W + (size_t)(kb + kk) * Nc + colB + ng),
                (sv4i*)(&w_lds[kk * 16 + ng]),
                0, 0);
#else
            *(float4*)&w_lds[kk * 16 + ng] =
                *(const float4*)&W[(size_t)(kb + kk) * Nc + colB + ng];
#endif
        }
#if HAVE_ASYNC_LDS
        __builtin_amdgcn_s_wait_asynccnt(0);
#endif
        __syncthreads();

        v16bf a = load_frag_a(
            [&](int m, int kk) { return X[(size_t)(rowB + m) * K + kb + kk]; }, lane);
        v16bf b = load_frag_b(
            [&](int kk, int n) { return w_lds[kk * 16 + n]; }, lane);
        acc = wmma_bf16(a, b, acc);
        __syncthreads();
    }
    const int n = lane & 15, half = lane >> 4;
#pragma unroll
    for (int j = 0; j < 8; ++j) {
        const int m   = (half << 3) + j;
        const int col = colB + n;
        float v = acc[j] + bias[col];
        if (act == 1) v = gelu_exact(v);
        Y[(size_t)(rowB + m) * Nc + col] = v;
    }
}

// ---------------------------------------------------------------------------
// Fused relative-position attention. One wave = one (n, h, 16-row l-block).
// Scores strip (16 x 512) + QE (16 x 257) + KT (16 x 32) live in LDS.
// ---------------------------------------------------------------------------
__global__ __launch_bounds__(32) void attention_kernel(
    const float* __restrict__ q, const float* __restrict__ k,
    const float* __restrict__ v, const float* __restrict__ emb,
    float* __restrict__ tout) {
    __shared__ float s_row[16 * 512];   // 32 KB: scores / probs for this l-block
    __shared__ float s_qe[16 * 257];    // 16 KB: QE[l, d] = q[l] . emb[d]
    __shared__ float s_ktd[16 * 32];    // 2 KB : KT[s, dd] = k[s] . emb[base+dd]

    const int lane = threadIdx.x & 31;
    const int lb   = blockIdx.x & 31;           // l-block (F/16 = 32)
    const int h    = (blockIdx.x >> 5) & 7;
    const int n    = blockIdx.x >> 8;
    const int lBase = lb << 4;
    const int hOff  = h * E;

    // ---- A fragment: Q block (16 x 32), reused everywhere -------------------
    v16bf aq = load_frag_a(
        [&](int m, int kk) {
            return q[((size_t)(lBase + m) * N + n) * C + hOff + kk];
        },
        lane);

    // ---- QE = Q @ emb^T  (16 x 257 in 17 column tiles) ----------------------
    for (int dt = 0; dt < 17; ++dt) {
        v8f c = {};
        v16bf be = load_frag_b(
            [&](int kk, int nn) {
                int col = dt * 16 + nn; if (col > 256) col = 256;
                return emb[(size_t)col * E + kk];
            },
            lane);
        c = wmma_bf16(aq, be, c);
        const int nn = lane & 15, half = lane >> 4;
#pragma unroll
        for (int j = 0; j < 8; ++j) {
            const int col = dt * 16 + nn;
            if (col < 257) s_qe[((half << 3) + j) * 257 + col] = c[j];
        }
    }

    // ---- score strip: loop s-tiles ------------------------------------------
    const float rsE = 0.17677669529663689f;  // 1/sqrt(32)
    for (int st = 0; st < 32; ++st) {
        const int sBase = st << 4;
        int base = lBase - sBase - 15;                     // min unclipped l-s in tile
        base = base < -MP ? -MP : (base > MP ? MP : base); // clip like pe does

        // KT[s_local, dd] = k[sBase+s_local] . emb[base+128+dd], dd in [0,32)
        v16bf ak = load_frag_a(
            [&](int m, int kk) {
                return k[((size_t)(sBase + m) * N + n) * C + hOff + kk];
            },
            lane);
#pragma unroll
        for (int eo = 0; eo < 2; ++eo) {
            v8f c = {};
            v16bf be = load_frag_b(
                [&](int kk, int nn) {
                    int idx = base + MP + eo * 16 + nn; if (idx > 2 * MP) idx = 2 * MP;
                    return emb[(size_t)idx * E + kk];
                },
                lane);
            c = wmma_bf16(ak, be, c);
            const int nn = lane & 15, half = lane >> 4;
#pragma unroll
            for (int j = 0; j < 8; ++j)
                s_ktd[((half << 3) + j) * 32 + eo * 16 + nn] = c[j];
        }

        // raw Q.K^T tile
        v8f c = {};
        v16bf bk = load_frag_b(
            [&](int kk, int nn) {
                return k[((size_t)(sBase + nn) * N + n) * C + hOff + kk];
            },
            lane);
        c = wmma_bf16(aq, bk, c);

        // epilogue: add gathered pe terms, scale, stash strip
        const int nn = lane & 15, half = lane >> 4;
#pragma unroll
        for (int j = 0; j < 8; ++j) {
            const int m = (half << 3) + j;
            const int l = lBase + m, s = sBase + nn;
            int di = l - s; di = di < -MP ? -MP : (di > MP ? MP : di);
            // KT row index is s_local (= nn), column dd = clip(l-s) - base
            const float t2 = s_qe[m * 257 + di + MP];
            const float t3 = s_ktd[nn * 32 + (di - base)];
            s_row[m * 512 + s] = (c[j] + t2 + t3) * rsE;
        }
    }
    __syncthreads();

    // ---- softmax over s (each half-wave owns 256 cols of row lane&15) -------
    {
        const int r = lane & 15, hb = lane >> 4, cb = hb << 8;
        float mx = -3.0e38f;
        for (int i = 0; i < 256; ++i) mx = fmaxf(mx, s_row[r * 512 + cb + i]);
        mx = fmaxf(mx, __shfl_xor(mx, 16));
        float sum = 0.0f;
        for (int i = 0; i < 256; ++i) {
            float e = expf(s_row[r * 512 + cb + i] - mx);
            s_row[r * 512 + cb + i] = e;
            sum += e;
        }
        sum += __shfl_xor(sum, 16);
        const float inv = 1.0f / sum;
        for (int i = 0; i < 256; ++i) s_row[r * 512 + cb + i] *= inv;
    }
    __syncthreads();

    // ---- O = att @ V  (16 x 32, K = 512 in chunks of 32) --------------------
    v8f o0 = {}, o1 = {};
    for (int ch = 0; ch < 16; ++ch) {
        v16bf aa = load_frag_a(
            [&](int m, int kk) { return s_row[m * 512 + ch * 32 + kk]; }, lane);
#pragma unroll
        for (int eo = 0; eo < 2; ++eo) {
            v16bf bv = load_frag_b(
                [&](int kk, int nn) {
                    return v[((size_t)(ch * 32 + kk) * N + n) * C + hOff + eo * 16 + nn];
                },
                lane);
            if (eo == 0) o0 = wmma_bf16(aa, bv, o0);
            else         o1 = wmma_bf16(aa, bv, o1);
        }
    }
    const int nn = lane & 15, half = lane >> 4;
#pragma unroll
    for (int j = 0; j < 8; ++j) {
        const int l = lBase + (half << 3) + j;
        tout[((size_t)l * N + n) * C + hOff + nn]      = o0[j];
        tout[((size_t)l * N + n) * C + hOff + 16 + nn] = o1[j];
    }
}

// ---------------------------------------------------------------------------
// Row kernels: gated AddNorm, and residual + LayerNorm
// ---------------------------------------------------------------------------
__device__ __forceinline__ float block_sum256(float vIn, float* red) {
    const int tid = threadIdx.x;
    red[tid] = vIn;
    __syncthreads();
    for (int s = 128; s > 0; s >>= 1) {
        if (tid < s) red[tid] += red[tid + s];
        __syncthreads();
    }
    const float r = red[0];
    __syncthreads();
    return r;
}

__global__ __launch_bounds__(256) void gate_ln1_kernel(
    const float* __restrict__ x, const float* __restrict__ t,
    const float* __restrict__ bw, const float* __restrict__ g,
    const float* __restrict__ b, float* __restrict__ h1) {
    __shared__ float red[256];
    const size_t row = blockIdx.x;
    const int c = threadIdx.x;
    const float xv = x[row * C + c], tv = t[row * C + c];
    const float contrib = tv * bw[c] + xv * bw[C + c] + (tv - xv) * bw[2 * C + c];
    const float gs = block_sum256(contrib, red);
    const float gate = 1.0f / (1.0f + expf(-gs));
    const float y = gate * xv + (1.0f - gate) * tv;
    const float mu = block_sum256(y, red) * (1.0f / C);
    const float d = y - mu;
    const float var = block_sum256(d * d, red) * (1.0f / C);
    h1[row * C + c] = d * rsqrtf(var + EPS) * g[c] + b[c];
}

__global__ __launch_bounds__(256) void add_ln_kernel(
    const float* __restrict__ a, const float* __restrict__ bsrc,
    const float* __restrict__ g, const float* __restrict__ be,
    float* __restrict__ out) {
    __shared__ float red[256];
    const size_t row = blockIdx.x;
    const int c = threadIdx.x;
    const float y = a[row * C + c] + bsrc[row * C + c];
    const float mu = block_sum256(y, red) * (1.0f / C);
    const float d = y - mu;
    const float var = block_sum256(d * d, red) * (1.0f / C);
    out[row * C + c] = d * rsqrtf(var + EPS) * g[c] + be[c];
}

// ---------------------------------------------------------------------------
// Conv1d (C->C) + eval BatchNorm, as implicit GEMM. One wave = 16 co x 16 f.
// transposed=1: input is h2 in (F,N,C) layout; else (N,C,Lin) contiguous.
// ---------------------------------------------------------------------------
__global__ __launch_bounds__(256) void conv_bn_kernel(
    const float* __restrict__ zin, int transposed, const float* __restrict__ w,
    const float* __restrict__ cb, const float* __restrict__ bg,
    const float* __restrict__ bb, const float* __restrict__ bm,
    const float* __restrict__ bvv, float* __restrict__ zout,
    int Kw, int stride, int pad, int Lin, int Lout) {
    const int lane   = threadIdx.x & 31;
    const int waveId = blockIdx.x * (blockDim.x >> 5) + (threadIdx.x >> 5);
    const int fTiles = Lout >> 4;
    const int tilesPerN = (C >> 4) * fTiles;
    const int nb = waveId / tilesPerN;
    const int r  = waveId % tilesPerN;
    const int coT = r / fTiles, fT = r % fTiles;
    if (nb >= N) return;

    __builtin_prefetch(w + (size_t)coT * 16 * C * Kw, 0, 3);

    v8f acc = {};
    for (int t = 0; t < Kw; ++t) {
        for (int cc = 0; cc < C; cc += 32) {
            v16bf a = load_frag_a(
                [&](int m, int kk) {
                    return w[((size_t)(coT * 16 + m) * C + cc + kk) * Kw + t];
                },
                lane);
            v16bf b = load_frag_b(
                [&](int kk, int nn) {
                    const int fpos = stride * (fT * 16 + nn) + t - pad;
                    if (fpos < 0 || fpos >= Lin) return 0.0f;
                    return transposed
                               ? zin[((size_t)fpos * N + nb) * C + cc + kk]
                               : zin[((size_t)nb * C + cc + kk) * Lin + fpos];
                },
                lane);
            acc = wmma_bf16(a, b, acc);
        }
    }
    const int nn = lane & 15, half = lane >> 4;
#pragma unroll
    for (int j = 0; j < 8; ++j) {
        const int co = coT * 16 + (half << 3) + j;
        const int f  = fT * 16 + nn;
        const float scale = bg[co] * rsqrtf(bvv[co] + EPS);
        zout[((size_t)nb * C + co) * Lout + f] =
            (acc[j] + cb[co] - bm[co]) * scale + bb[co];
    }
}

// ---------------------------------------------------------------------------
// Host-side orchestration
// ---------------------------------------------------------------------------
extern "C" void kernel_launch(void* const* d_in, const int* in_sizes, int n_in,
                              void* d_out, int out_size, void* d_ws, size_t ws_size,
                              hipStream_t stream) {
    (void)in_sizes; (void)n_in; (void)out_size; (void)ws_size;
    const float* x    = (const float*)d_in[0];
    const float* Wq   = (const float*)d_in[1];
    const float* bq   = (const float*)d_in[2];
    const float* Wk   = (const float*)d_in[3];
    const float* bk   = (const float*)d_in[4];
    const float* Wv   = (const float*)d_in[5];
    const float* bv   = (const float*)d_in[6];
    const float* emb  = (const float*)d_in[7];
    const float* bw   = (const float*)d_in[8];
    const float* ln1g = (const float*)d_in[9];
    const float* ln1b = (const float*)d_in[10];
    const float* w1   = (const float*)d_in[11];
    const float* b1   = (const float*)d_in[12];
    const float* w2   = (const float*)d_in[13];
    const float* b2   = (const float*)d_in[14];
    const float* ln2g = (const float*)d_in[15];
    const float* ln2b = (const float*)d_in[16];
    const float* cw0 = (const float*)d_in[17]; const float* cb0 = (const float*)d_in[18];
    const float* g0  = (const float*)d_in[19]; const float* be0 = (const float*)d_in[20];
    const float* m0  = (const float*)d_in[21]; const float* v0  = (const float*)d_in[22];
    const float* cw1 = (const float*)d_in[23]; const float* cb1 = (const float*)d_in[24];
    const float* g1  = (const float*)d_in[25]; const float* be1 = (const float*)d_in[26];
    const float* m1  = (const float*)d_in[27]; const float* v1  = (const float*)d_in[28];
    const float* cw2 = (const float*)d_in[29]; const float* cb2 = (const float*)d_in[30];
    const float* g2  = (const float*)d_in[31]; const float* be2 = (const float*)d_in[32];
    const float* m2  = (const float*)d_in[33]; const float* v2  = (const float*)d_in[34];

    float* ws = (float*)d_ws;
    const size_t SZ = (size_t)F * N * C;            // 4 Mi floats
    float* q   = ws + 0 * SZ;
    float* k   = ws + 1 * SZ;
    float* v   = ws + 2 * SZ;
    float* t   = ws + 3 * SZ;
    float* h1  = ws + 4 * SZ;
    float* mid = ws + 0 * SZ;   // q dead after attention
    float* ff  = ws + 1 * SZ;   // k dead
    float* h2  = ws + 2 * SZ;   // v dead
    float* z0  = ws + 3 * SZ;   // t dead after gate_ln1
    float* z1  = ws + 0 * SZ;   // mid dead after gemm2

    const int M = F * N;                 // 16384 rows
    const int gemmBlocks = (M / 128) * (C / 16);      // 2048 blocks (8 waves in M)

    // QKV projections
    gemm_bias_act_kernel<<<gemmBlocks, 256, 0, stream>>>(x, Wq, bq, q, M, C, C, 0);
    gemm_bias_act_kernel<<<gemmBlocks, 256, 0, stream>>>(x, Wk, bk, k, M, C, C, 0);
    gemm_bias_act_kernel<<<gemmBlocks, 256, 0, stream>>>(x, Wv, bv, v, M, C, C, 0);

    // Fused relative-position attention (N*H*(F/16) waves)
    attention_kernel<<<N * H * (F / 16), 32, 0, stream>>>(q, k, v, emb, t);

    // Gated AddNorm -> h1
    gate_ln1_kernel<<<M, 256, 0, stream>>>(x, t, bw, ln1g, ln1b, h1);

    // FFN: gelu(h1 @ w1 + b1) @ w2 + b2, then residual + LN2 -> h2
    gemm_bias_act_kernel<<<gemmBlocks, 256, 0, stream>>>(h1, w1, b1, mid, M, C, C, 1);
    gemm_bias_act_kernel<<<gemmBlocks, 256, 0, stream>>>(mid, w2, b2, ff, M, C, C, 0);
    add_ln_kernel<<<M, 256, 0, stream>>>(ff, h1, ln2g, ln2b, h2);

    // Temporal convs (implicit GEMM + BN)
    conv_bn_kernel<<<N * (C / 16) * (F / 16) / 8, 256, 0, stream>>>(
        h2, 1, cw0, cb0, g0, be0, m0, v0, z0, 9, 1, 4, F, F);
    conv_bn_kernel<<<N * (C / 16) * (F / 16) / 8, 256, 0, stream>>>(
        z0, 0, cw1, cb1, g1, be1, m1, v1, z1, 9, 1, 4, F, F);
    conv_bn_kernel<<<N * (C / 16) * (F / 32) / 8, 256, 0, stream>>>(
        z1, 0, cw2, cb2, g2, be2, m2, v2, (float*)d_out, 5, 2, 2, F, F / 2);
}
